// MambaShared_33492154974300
// MI455X (gfx1250) — matrix operand order repaired and checked by reading, not compiled, not run on hardware
//
#include <hip/hip_runtime.h>
#include <cstdint>
#include <cstddef>

// ---------------- problem dims (match reference) ----------------
constexpr int kBatch  = 2;
constexpr int kSeq    = 1024;
constexpr int kDm     = 1024;
constexpr int kDstate = 64;
constexpr int kDconv  = 4;
constexpr int kDin    = 2048;            // EXPAND * DM
constexpr int kDtr    = 64;              // (DM+15)/16
constexpr int kLayers = 6;
constexpr int kXdbl   = kDtr + 2 * kDstate;   // 192
constexpr int kRows   = kBatch * kSeq;        // 2048 (flattened b,l)
constexpr float kEps  = 1e-5f;

typedef __attribute__((ext_vector_type(16))) __bf16 bf16x16;
typedef __attribute__((ext_vector_type(8)))  float  floatx8;
typedef unsigned short u16;
typedef unsigned int   u32;

__device__ __forceinline__ u16 f2bf(float f) {
  u32 u = __float_as_uint(f);
  u += 0x7FFFu + ((u >> 16) & 1u);           // round-to-nearest-even
  return (u16)(u >> 16);
}
__device__ __forceinline__ float sigm(float x) { return 1.0f / (1.0f + __expf(-x)); }

// ---- gfx1250 async global->LDS path (ASYNCcnt), with sync fallback ----
#define HAS_ASYNC_LDS __has_builtin(__builtin_amdgcn_global_load_async_to_lds_b128)

#if HAS_ASYNC_LDS
// Builtin expects: (int4 AS1* global_src, int4 AS3* lds_dst, imm offset, imm cpol)
typedef int v4i_t __attribute__((vector_size(16)));
typedef __attribute__((address_space(1))) v4i_t* gptr_b128;
typedef __attribute__((address_space(3))) v4i_t* lptr_b128;
__device__ __forceinline__ void async_copy_b128(const void* g, void* l) {
  __builtin_amdgcn_global_load_async_to_lds_b128((gptr_b128)g, (lptr_b128)l, 0, 0);
}
__device__ __forceinline__ void wait_async0() {
#if __has_builtin(__builtin_amdgcn_s_wait_asynccnt)
  __builtin_amdgcn_s_wait_asynccnt(0);
#else
  asm volatile("s_wait_asynccnt 0x0" ::: "memory");
#endif
}
#endif

// ---------------- f32 -> bf16 conversion (weights) ----------------
__global__ void cvt_f32_to_bf16(const float* __restrict__ s, u16* __restrict__ d, int n) {
  int i = blockIdx.x * blockDim.x + threadIdx.x;
  int stride = gridDim.x * blockDim.x;
  for (; i < n; i += stride) d[i] = f2bf(s[i]);
}

// ---------------- LayerNorm -> bf16 (one block per row of 1024) ----------------
__global__ void __launch_bounds__(256) layernorm_bf16(
    const float* __restrict__ x, const float* __restrict__ w,
    const float* __restrict__ b, u16* __restrict__ out) {
  const int row = blockIdx.x;
  const int tid = threadIdx.x;
  const float4 v = ((const float4*)(x + (size_t)row * kDm))[tid];
  __shared__ float s1[256], s2[256];
  s1[tid] = v.x + v.y + v.z + v.w;
  s2[tid] = v.x * v.x + v.y * v.y + v.z * v.z + v.w * v.w;
  __syncthreads();
  for (int o = 128; o > 0; o >>= 1) {
    if (tid < o) { s1[tid] += s1[tid + o]; s2[tid] += s2[tid + o]; }
    __syncthreads();
  }
  const float mu = s1[0] * (1.0f / kDm);
  const float var = s2[0] * (1.0f / kDm) - mu * mu;
  const float rs = rsqrtf(var + kEps);
  const float4 wv = ((const float4*)w)[tid];
  const float4 bv = ((const float4*)b)[tid];
  u16* o4 = out + (size_t)row * kDm + tid * 4;
  o4[0] = f2bf((v.x - mu) * rs * wv.x + bv.x);
  o4[1] = f2bf((v.y - mu) * rs * wv.y + bv.y);
  o4[2] = f2bf((v.z - mu) * rs * wv.z + bv.z);
  o4[3] = f2bf((v.w - mu) * rs * wv.w + bv.w);
}

// ---------------- depthwise causal conv (k=4) + bias + SiLU ----------------
__global__ void conv_silu(const float* __restrict__ xz, const float* __restrict__ cw,
                          const float* __restrict__ cb, u16* __restrict__ xc_bf,
                          float* __restrict__ xc_f) {
  const int idx = blockIdx.x * blockDim.x + threadIdx.x;   // over kRows*kDin
  const int d  = idx & (kDin - 1);
  const int l  = (idx / kDin) & (kSeq - 1);
  const int bl = idx / kDin;                                // b*kSeq + l
  float acc = cb[d];
#pragma unroll
  for (int j = 0; j < kDconv; j++) {
    const int ll = l - (kDconv - 1) + j;
    if (ll >= 0)
      acc += cw[d * kDconv + j] * xz[(size_t)(bl - (kDconv - 1) + j) * (2 * kDin) + d];
  }
  const float sv = acc * sigm(acc);
  xc_f[idx]  = sv;
  xc_bf[idx] = f2bf(sv);
}

// ---------------- extract dt columns of x_dbl as bf16 (lda = kDtr) ----------------
__global__ void extract_dt_bf16(const float* __restrict__ xdbl, u16* __restrict__ dtb) {
  const int idx = blockIdx.x * blockDim.x + threadIdx.x;    // kRows * kDtr
  const int r = idx >> 6, c = idx & 63;
  dtb[idx] = f2bf(xdbl[(size_t)r * kXdbl + c]);
}

// ---------------- WMMA bf16 GEMM: C[M,N] = A[M,K] * W[N,K]^T (+ epilogue) ------
// Block tile 128(M) x 64(N), KT=32, 8 waves each computing a 32x32 region (2x2 WMMA).
// Double-buffered LDS; tiles staged with GLOBAL_LOAD_ASYNC_TO_LDS_B128 (ASYNCcnt)
// so the k+1 copy overlaps the k-tile WMMAs. EPI: 0 = store; 1 = += residual;
// 2 = softplus(v + bias[col]).
template <int EPI>
__global__ void __launch_bounds__(256) gemm_wmma(
    const u16* __restrict__ A, int lda, const u16* __restrict__ W, int ldw,
    float* __restrict__ C, int ldc, const float* __restrict__ aux,
    int M, int N, int K) {
  __shared__ u16 sA[2][128 * 32];   // [m][k] row-major
  __shared__ u16 sW[2][64 * 32];    // [n][k] row-major  (B^T staged rows)
  const int tid  = threadIdx.x;
  const int lane = tid & 31;
  const int wave = tid >> 5;
  const int lm = lane & 15;
  const int lh = lane >> 4;
  const int mBase = blockIdx.y * 128;
  const int nBase = blockIdx.x * 64;
  const int waveM = (wave >> 1) * 32;
  const int waveN = (wave & 1) * 32;

  // cooperative copy assignments (uint4 = 8 bf16):
  //   A tile: 512 uint4 -> 2 per thread (rows r and r+64); W tile: 256 uint4 -> 1
  const int r0 = tid >> 2, g8 = (tid & 3) * 8;

  const floatx8 vz = {0, 0, 0, 0, 0, 0, 0, 0};
  floatx8 acc[2][2] = {{vz, vz}, {vz, vz}};

  auto issue_tile = [&](int buf, int k0) {
    const u16* ga0 = A + (size_t)(mBase + r0) * lda + k0 + g8;
    const u16* ga1 = A + (size_t)(mBase + 64 + r0) * lda + k0 + g8;
    const u16* gw  = W + (size_t)(nBase + r0) * ldw + k0 + g8;
    u16* la0 = &sA[buf][(size_t)tid * 8];
    u16* la1 = &sA[buf][(size_t)(tid + 256) * 8];
    u16* lw  = &sW[buf][(size_t)tid * 8];
#if HAS_ASYNC_LDS
    async_copy_b128(ga0, la0);
    async_copy_b128(ga1, la1);
    async_copy_b128(gw, lw);
#else
    *(uint4*)la0 = *(const uint4*)ga0;
    *(uint4*)la1 = *(const uint4*)ga1;
    *(uint4*)lw  = *(const uint4*)gw;
#endif
    if (k0 + 32 < K) {  // gfx1250 global_prefetch_b8 for the tile after next
      __builtin_prefetch(ga0 + 32, 0, 1);
      __builtin_prefetch(gw + 32, 0, 1);
    }
  };

  issue_tile(0, 0);
  int cb = 0;
  for (int k0 = 0; k0 < K; k0 += 32, cb ^= 1) {
#if HAS_ASYNC_LDS
    wait_async0();          // this wave's async writes to LDS have landed
#endif
    __syncthreads();        // all waves' tile writes visible; prior reads done
    if (k0 + 32 < K) issue_tile(cb ^ 1, k0 + 32);   // overlap with compute

    const u32* sA32 = (const u32*)sA[cb];   // [m][k/2], 16 dwords per row
    const u32* sW32 = (const u32*)sW[cb];   // [n][k/2]
    union Frag { bf16x16 v; u32 u[8]; };
    Frag af[2], bfr[2];
    // A fragment: 16-bit A 16x32 layout — VGPR p holds K pair
    //   k = (p/4)*16 + (lane/16)*8 + (p%4)*2  (pairs k,k+1 -> one dword)
#pragma unroll
    for (int i = 0; i < 2; i++) {
      const int mT = waveM + i * 16 + lm;
#pragma unroll
      for (int p = 0; p < 8; p++) {
        const int k2 = ((p >> 2) << 3) + (lh << 2) + (p & 3);   // (k/2)
        af[i].u[p] = sA32[mT * 16 + k2];
      }
    }
    // B fragment (32x16, same K order as A):
    //   VGPR p holds K pair k = (p/4)*16 + (lane/16)*8 + (p%4)*2; col = lane%16
#pragma unroll
    for (int j = 0; j < 2; j++) {
      const int nT = waveN + j * 16 + lm;
#pragma unroll
      for (int p = 0; p < 8; p++)
        bfr[j].u[p] = sW32[nT * 16 + ((p >> 2) << 3) + (lh << 2) + (p & 3)];
    }
#pragma unroll
    for (int i = 0; i < 2; i++)
#pragma unroll
      for (int j = 0; j < 2; j++)
        acc[i][j] = __builtin_amdgcn_wmma_f32_16x16x32_bf16(
            false, af[i].v, false, bfr[j].v, (short)0, acc[i][j], false, false);
  }

  // epilogue: C/D layout — VGPR r: lanes 0-15 M=r, lanes 16-31 M=r+8; N = lane%16
#pragma unroll
  for (int i = 0; i < 2; i++) {
#pragma unroll
    for (int j = 0; j < 2; j++) {
      const int col = nBase + waveN + j * 16 + lm;
#pragma unroll
      for (int r = 0; r < 8; r++) {
        const int row = mBase + waveM + i * 16 + r + lh * 8;
        float v = acc[i][j][r];
        if (EPI == 1) v += aux[(size_t)row * ldc + col];
        if (EPI == 2) { v += aux[col]; v = (v > 20.f) ? v : log1pf(__expf(v)); }
        C[(size_t)row * ldc + col] = v;
      }
    }
  }
}

// ---------------- selective scan, fused with y * silu(z) -> bf16 ----------------
// thread = one (b, d) channel; h[64], A[d][64] live in VGPRs; B/C broadcast via LDS
__global__ void __launch_bounds__(256) scan_kernel(
    const float* __restrict__ u, const float* __restrict__ delta,
    const float* __restrict__ xdbl, const float* __restrict__ A_log,
    const float* __restrict__ Dv, const float* __restrict__ xz,
    u16* __restrict__ ybf) {
  const int d = blockIdx.x * 256 + threadIdx.x;
  const int b = blockIdx.y;
  const int tid = threadIdx.x;
  float Ad[kDstate], h[kDstate];
#pragma unroll
  for (int n = 0; n < kDstate; n++) {
    Ad[n] = -__expf(A_log[(size_t)d * kDstate + n]);
    h[n] = 0.f;
  }
  const float Dd = Dv[d];
  __shared__ float sB[kDstate], sC[kDstate];
  for (int t = 0; t < kSeq; t++) {
    const size_t bl = (size_t)b * kSeq + t;
    if (tid < kDstate)
      sB[tid] = xdbl[bl * kXdbl + kDtr + tid];
    else if (tid < 2 * kDstate)
      sC[tid - kDstate] = xdbl[bl * kXdbl + kDtr + kDstate + (tid - kDstate)];
    __syncthreads();
    const float dl = delta[bl * kDin + d];
    const float uv = u[bl * kDin + d];
    const float du = dl * uv;
    float y = 0.f;
#pragma unroll
    for (int n = 0; n < kDstate; n++) {
      const float dA = __expf(dl * Ad[n]);
      h[n] = dA * h[n] + du * sB[n];
      y += h[n] * sC[n];
    }
    y += uv * Dd;
    const float z = xz[bl * (2 * kDin) + kDin + d];
    y *= z * sigm(z);
    ybf[bl * kDin + d] = f2bf(y);
    __syncthreads();
  }
}

// ---------------- host orchestration ----------------
extern "C" void kernel_launch(void* const* d_in, const int* in_sizes, int n_in,
                              void* d_out, int out_size, void* d_ws, size_t ws_size,
                              hipStream_t stream) {
  (void)in_sizes; (void)n_in; (void)out_size; (void)ws_size;
  const float* x_in   = (const float*)d_in[0];
  const float* in_w   = (const float*)d_in[1];
  const float* conv_w = (const float*)d_in[2];
  const float* conv_b = (const float*)d_in[3];
  const float* xp_w   = (const float*)d_in[4];
  const float* dt_w   = (const float*)d_in[5];
  const float* dt_b   = (const float*)d_in[6];
  const float* A_log  = (const float*)d_in[7];
  const float* Dp     = (const float*)d_in[8];
  const float* out_w  = (const float*)d_in[9];
  const float* nw     = (const float*)d_in[10];
  const float* nb     = (const float*)d_in[11];
  float* out_final = (float*)d_out;

  char* ws = (char*)d_ws;
  size_t off = 0;
  auto alloc = [&](size_t bytes) -> char* {
    char* p = ws + off;
    off = (off + bytes + 255) & ~(size_t)255;
    return p;
  };
  u16*   w_in_bf  = (u16*)alloc((size_t)2 * 2 * kDin * kDm * 2);
  u16*   w_xp_bf  = (u16*)alloc((size_t)2 * kXdbl * kDin * 2);
  u16*   w_dt_bf  = (u16*)alloc((size_t)2 * kDin * kDtr * 2);
  u16*   w_out_bf = (u16*)alloc((size_t)2 * kDm * kDin * 2);
  u16*   h_bf     = (u16*)alloc((size_t)kRows * kDm * 2);
  float* xzbuf    = (float*)alloc((size_t)kRows * 2 * kDin * 4);
  u16*   xc_bf    = (u16*)alloc((size_t)kRows * kDin * 2);
  float* xc_f     = (float*)alloc((size_t)kRows * kDin * 4);
  float* xdbl     = (float*)alloc((size_t)kRows * kXdbl * 4);
  u16*   dt_bf    = (u16*)alloc((size_t)kRows * kDtr * 2);
  float* delta    = (float*)alloc((size_t)kRows * kDin * 4);
  u16*   y_bf     = (u16*)alloc((size_t)kRows * kDin * 2);
  float* xA       = (float*)alloc((size_t)kRows * kDm * 4);
  float* xB       = (float*)alloc((size_t)kRows * kDm * 4);

  auto cvt = [&](const float* s, u16* dst, size_t n) {
    int blocks = (int)((n + 255) / 256);
    if (blocks > 16384) blocks = 16384;
    cvt_f32_to_bf16<<<blocks, 256, 0, stream>>>(s, dst, (int)n);
  };
  cvt(in_w,  w_in_bf,  (size_t)2 * 2 * kDin * kDm);
  cvt(xp_w,  w_xp_bf,  (size_t)2 * kXdbl * kDin);
  cvt(dt_w,  w_dt_bf,  (size_t)2 * kDin * kDtr);
  cvt(out_w, w_out_bf, (size_t)2 * kDm * kDin);

  const float* cur = x_in;
  for (int layer = 0; layer < kLayers; layer++) {
    const int blk = (layer < 3) ? 0 : 1;
    float* nxt = (layer == kLayers - 1) ? out_final : ((layer & 1) ? xB : xA);

    layernorm_bf16<<<kRows, 256, 0, stream>>>(
        cur, nw + (size_t)layer * kDm, nb + (size_t)layer * kDm, h_bf);

    {  // in_proj: [2048 x 4096] = h_bf[2048 x 1024] * in_w^T
      dim3 g((2 * kDin) / 64, kRows / 128);
      gemm_wmma<0><<<g, 256, 0, stream>>>(
          h_bf, kDm, w_in_bf + (size_t)blk * 2 * kDin * kDm, kDm,
          xzbuf, 2 * kDin, nullptr, kRows, 2 * kDin, kDm);
    }
    conv_silu<<<(kRows * kDin) / 256, 256, 0, stream>>>(
        xzbuf, conv_w + (size_t)blk * kDin * kDconv,
        conv_b + (size_t)blk * kDin, xc_bf, xc_f);
    {  // x_proj: [2048 x 192] = xc[2048 x 2048] * xp_w^T
      dim3 g(kXdbl / 64, kRows / 128);
      gemm_wmma<0><<<g, 256, 0, stream>>>(
          xc_bf, kDin, w_xp_bf + (size_t)blk * kXdbl * kDin, kDin,
          xdbl, kXdbl, nullptr, kRows, kXdbl, kDin);
    }
    extract_dt_bf16<<<(kRows * kDtr) / 256, 256, 0, stream>>>(xdbl, dt_bf);
    {  // delta: softplus([2048 x 2048] = dt[2048 x 64] * dt_w^T + dt_b)
      dim3 g(kDin / 64, kRows / 128);
      gemm_wmma<2><<<g, 256, 0, stream>>>(
          dt_bf, kDtr, w_dt_bf + (size_t)blk * kDin * kDtr, kDtr,
          delta, kDin, dt_b + (size_t)blk * kDin, kRows, kDin, kDtr);
    }
    {  // selective scan + gate
      dim3 g(kDin / 256, kBatch);
      scan_kernel<<<g, 256, 0, stream>>>(
          xc_f, delta, xdbl, A_log + (size_t)blk * kDin * kDstate,
          Dp + (size_t)blk * kDin, xzbuf, y_bf);
    }
    {  // out_proj + residual: nxt = y * out_w^T + cur
      dim3 g(kDm / 64, kRows / 128);
      gemm_wmma<1><<<g, 256, 0, stream>>>(
          y_bf, kDin, w_out_bf + (size_t)blk * kDm * kDin, kDin,
          nxt, kDm, cur, kRows, kDm, kDin);
    }
    cur = nxt;
  }
}
